// Block_3702261809700
// MI455X (gfx1250) — matrix operand order repaired and checked by reading, not compiled, not run on hardware
//
#include <hip/hip_runtime.h>
#include <math.h>
#include <stdint.h>

// Problem constants (from reference)
#define Bz 2
#define Tz 2048
#define Ez 1024
#define Hz 16
#define Dz 64
#define Mz (Bz * Tz)      // 4096 token rows
#define FF (4 * Ez)       // 4096 MLP hidden

typedef __attribute__((ext_vector_type(16))) __bf16 v16bf;
typedef __attribute__((ext_vector_type(8)))  float  v8f;
typedef __attribute__((ext_vector_type(4)))  unsigned int uint4v;

union Frag { v16bf v; uint4v q[2]; };

// ---------- helpers ----------
__device__ __forceinline__ __bf16 f2bf(float f) {
  unsigned int u = __float_as_uint(f);
  u += 0x7FFFu + ((u >> 16) & 1u);              // round-to-nearest-even
  unsigned short h = (unsigned short)(u >> 16);
  return __builtin_bit_cast(__bf16, h);
}

// Async global -> LDS copy of 16B per lane (CDNA5, tracked by ASYNCcnt)
__device__ __forceinline__ void async_copy_b128(unsigned ldsOff, const void* gaddr) {
  asm volatile("global_load_async_to_lds_b128 %0, %1, off"
               :: "v"(ldsOff), "v"(gaddr) : "memory");
}

// A-matrix 16x32 bf16 fragment, K contiguous in memory, sm elems between rows.
// lanes 0-15: m=lane, elems = K 8h..8h+7 then 16+8h..16+8h+7 (h = lane>>4)
__device__ __forceinline__ v16bf load_A_frag_k1(const __bf16* base, int sm) {
  int lane = threadIdx.x & 31;
  int half = lane >> 4, m = lane & 15;
  Frag f;
  f.q[0] = *(const uint4v*)(base + m * sm + 8 * half);
  f.q[1] = *(const uint4v*)(base + m * sm + 16 + 8 * half);
  return f.v;
}

// B-matrix 32x16 bf16 fragment, K contiguous in memory, sn elems between cols.
// lane: n = lane&15; elems = K 8h..8h+7 then 16+8h..16+8h+7 (h = lane>>4)
__device__ __forceinline__ v16bf load_B_frag_k1(const __bf16* base, int sn) {
  int lane = threadIdx.x & 31;
  int khalf = lane >> 4, n = lane & 15;
  Frag f;
  f.q[0] = *(const uint4v*)(base + n * sn + 8 * khalf);
  f.q[1] = *(const uint4v*)(base + n * sn + 16 + 8 * khalf);
  return f.v;
}

// ---------- fp32 weights -> transposed bf16 Wt[n*K + k] ----------
__global__ void cvt_transpose_kernel(const float* __restrict__ W,
                                     __bf16* __restrict__ Wt,
                                     int kshift, int N) {
  size_t i = (size_t)blockIdx.x * 256 + threadIdx.x;   // linear over output
  int K = 1 << kshift;
  int k = (int)(i & (size_t)(K - 1));
  size_t n = i >> kshift;
  Wt[i] = f2bf(W[(size_t)k * N + n]);
}

// ---------- V transpose: [B*T, E](h-interleaved) -> vt[b,h,d,t] ----------
__global__ void transpose_v_kernel(const __bf16* __restrict__ vbf,
                                   __bf16* __restrict__ vt) {
  size_t i = (size_t)blockIdx.x * 256 + threadIdx.x;   // over B*H*D*T
  int t = (int)(i & (Tz - 1));
  size_t r = i >> 11;
  int d = (int)(r & (Dz - 1));
  size_t bh = r >> 6;
  int h = (int)(bh & (Hz - 1));
  int b = (int)(bh >> 4);
  vt[i] = vbf[((size_t)(b * Tz + t)) * Ez + h * Dz + d];
}

// ---------- LayerNorm (one block per row) -> bf16 ----------
__global__ __launch_bounds__(256) void ln_bf16_kernel(
    const float* __restrict__ x, const float* __restrict__ g,
    const float* __restrict__ bta, __bf16* __restrict__ out) {
  __shared__ float s1[256], s2[256];
  int r = blockIdx.x, tid = threadIdx.x;
  const float* xr = x + (size_t)r * Ez;
  float a = 0.f, b = 0.f;
  for (int i = tid; i < Ez; i += 256) { float v = xr[i]; a += v; b += v * v; }
  s1[tid] = a; s2[tid] = b;
  __syncthreads();
  for (int off = 128; off > 0; off >>= 1) {
    if (tid < off) { s1[tid] += s1[tid + off]; s2[tid] += s2[tid + off]; }
    __syncthreads();
  }
  float mu  = s1[0] * (1.0f / Ez);
  float var = s2[0] * (1.0f / Ez) - mu * mu;
  float rs  = rsqrtf(var + 1e-5f);
  __bf16* orow = out + (size_t)r * Ez;
  for (int i = tid; i < Ez; i += 256)
    orow[i] = f2bf((xr[i] - mu) * rs * g[i] + bta[i]);
}

// ---------- tiled WMMA GEMM: C[M,N] = A[M,K](bf16) * Wt[N,K](bf16, transposed) ----
// 128x128 block tile, 8 waves (4x2), each wave 32x64 = 2x4 wmma tiles, K step 32.
// Double-buffered LDS filled by async global->LDS b128 copies (ASYNCcnt).
template <int Kc, bool OUTF, bool OUTB, bool RELU, bool RES, bool BIAS, bool SCALE>
__global__ __launch_bounds__(256) void gemm_bf16_kernel(
    const __bf16* __restrict__ A, const __bf16* __restrict__ Wt,
    const float* __restrict__ bias, const float* __restrict__ resid,
    float* __restrict__ outF, __bf16* __restrict__ outB, int N) {
  __shared__ alignas(16) __bf16 As[2][128 * 32];   // [buf][m][k]
  __shared__ alignas(16) __bf16 Bs[2][128 * 32];   // [buf][n][k]
  int tid = threadIdx.x;
  int lane = tid & 31, wid = tid >> 5;
  int wm = wid & 3, wn = wid >> 2;                 // 4 M-waves x 2 N-waves
  int blkN = blockIdx.x * 128, blkM = blockIdx.y * 128;

  v8f zero = {};
  v8f acc[2][4];
#pragma unroll
  for (int i = 0; i < 2; ++i)
#pragma unroll
    for (int j = 0; j < 4; ++j) acc[i][j] = zero;

  constexpr int Kq = Kc / 8;                       // uint4 per K-row
  // staging: thread covers rows (tid>>2) and (tid>>2)+64, quad (tid&3)
  const uint4v* aPtr = (const uint4v*)A  + (size_t)(blkM + (tid >> 2)) * Kq + (tid & 3);
  const uint4v* bPtr = (const uint4v*)Wt + (size_t)(blkN + (tid >> 2)) * Kq + (tid & 3);
  unsigned aLds = (unsigned)(size_t)((uint4v*)&As[0][0] + tid);
  unsigned bLds = (unsigned)(size_t)((uint4v*)&Bs[0][0] + tid);

  auto stage = [&](int buf) __attribute__((always_inline)) {
#pragma unroll
    for (int i = 0; i < 2; ++i)
      async_copy_b128(aLds + buf * 8192 + i * 4096, aPtr + (size_t)i * 64 * Kq);
#pragma unroll
    for (int i = 0; i < 2; ++i)
      async_copy_b128(bLds + buf * 8192 + i * 4096, bPtr + (size_t)i * 64 * Kq);
    aPtr += 4; bPtr += 4;                          // advance 32 bf16 along K
  };

  stage(0);
  int buf = 0;
  for (int k0 = 0; k0 < Kc; k0 += 32, buf ^= 1) {
    asm volatile("s_wait_asynccnt 0" ::: "memory");  // own async copies landed
    __syncthreads();                                 // all waves' copies landed
    if (k0 + 32 < Kc) stage(buf ^ 1);                // overlap next tile with compute

    v16bf aF[2], bF[4];
#pragma unroll
    for (int mt = 0; mt < 2; ++mt)
      aF[mt] = load_A_frag_k1(&As[buf][(wm * 32 + mt * 16) * 32], 32);
#pragma unroll
    for (int nt = 0; nt < 4; ++nt)
      bF[nt] = load_B_frag_k1(&Bs[buf][(wn * 64 + nt * 16) * 32], 32);
#pragma unroll
    for (int mt = 0; mt < 2; ++mt)
#pragma unroll
      for (int nt = 0; nt < 4; ++nt)
        acc[mt][nt] = __builtin_amdgcn_wmma_f32_16x16x32_bf16(
            false, aF[mt], false, bF[nt], (short)0, acc[mt][nt], false, false);
  }

  // epilogue: scale -> bias -> relu -> residual (compile-time flags)
  int half = lane >> 4, ncol = lane & 15;
#pragma unroll
  for (int mt = 0; mt < 2; ++mt)
#pragma unroll
    for (int nt = 0; nt < 4; ++nt)
#pragma unroll
      for (int v = 0; v < 8; ++v) {
        int row = blkM + wm * 32 + mt * 16 + v + 8 * half;
        int col = blkN + wn * 64 + nt * 16 + ncol;
        float val = acc[mt][nt][v];
        if constexpr (SCALE) val *= 0.125f;      // 1/sqrt(D) folded into Q proj
        if constexpr (BIAS)  val += bias[col];
        if constexpr (RELU)  val = fmaxf(val, 0.f);
        if constexpr (RES)   val += resid[(size_t)row * N + col];
        if constexpr (OUTF)  outF[(size_t)row * N + col] = val;
        if constexpr (OUTB)  outB[(size_t)row * N + col] = f2bf(val);
      }
}

// ---------- flash attention: 1 wave per (b,h, 16-query tile) ----------
// Q (pre-scaled by 1/sqrt(D)), K in [B*T, E] head-interleaved bf16;
// V pre-transposed vt[b,h,d,t] bf16.
__global__ __launch_bounds__(256) void attn_fwd_kernel(
    const __bf16* __restrict__ Q, const __bf16* __restrict__ Kc,
    const __bf16* __restrict__ Vt, __bf16* __restrict__ O) {
  __shared__ alignas(16) __bf16 Plds[8][16 * 32];   // per-wave P scratch
  int wid = threadIdx.x >> 5, lane = threadIdx.x & 31;
  int widx = blockIdx.x * 8 + wid;                  // 4096 waves total
  int qt = widx >> 5;                               // / (B*H)=32
  int bh = widx & 31;
  int bb = bh >> 4, hh = bh & 15;
  int q0 = qt * 16;
  int half = lane >> 4, lm = lane & 15;
  const int base = bb * Tz * Ez + hh * Dz;
  const __bf16* vtp = Vt + ((size_t)(bb * Hz + hh) * Dz) * Tz;

  v16bf qa0 = load_A_frag_k1(Q + base + (size_t)q0 * Ez, Ez);       // d 0..31
  v16bf qa1 = load_A_frag_k1(Q + base + (size_t)q0 * Ez + 32, Ez);  // d 32..63

  v8f zero = {};
  v8f o_acc[4];
#pragma unroll
  for (int i = 0; i < 4; ++i) o_acc[i] = zero;
  float mrun[8], lrun[8];
#pragma unroll
  for (int v = 0; v < 8; ++v) { mrun[v] = -INFINITY; lrun[v] = 0.f; }

  __bf16* pl = &Plds[wid][0];

  auto process_tile = [&](int key0, bool domask) __attribute__((always_inline)) {
    v8f S[2] = {zero, zero};
#pragma unroll
    for (int t = 0; t < 2; ++t) {
      const __bf16* kp = Kc + base + (size_t)(key0 + 16 * t) * Ez;
      v16bf b0 = load_B_frag_k1(kp, Ez);            // contraction over d 0..31
      S[t] = __builtin_amdgcn_wmma_f32_16x16x32_bf16(
          false, qa0, false, b0, (short)0, S[t], false, false);
      v16bf b1 = load_B_frag_k1(kp + 32, Ez);       // d 32..63
      S[t] = __builtin_amdgcn_wmma_f32_16x16x32_bf16(
          false, qa1, false, b1, (short)0, S[t], false, false);
    }
    // online softmax (row = v + 8*half, col = lm); mask only on boundary tile
#pragma unroll
    for (int v = 0; v < 8; ++v) {
      if (domask) {
        int qg = q0 + v + 8 * half;
#pragma unroll
        for (int t = 0; t < 2; ++t) {
          int kg = key0 + 16 * t + lm;
          S[t][v] = (kg <= qg) ? S[t][v] : -INFINITY;
        }
      }
      float r = fmaxf(S[0][v], S[1][v]);
#pragma unroll
      for (int off = 8; off > 0; off >>= 1)
        r = fmaxf(r, __shfl_xor(r, off, 32));       // reduce within 16-lane group
      float mnew = fmaxf(mrun[v], r);
      float corr = __expf(mrun[v] - mnew);
      float p0 = __expf(S[0][v] - mnew);
      float p1 = __expf(S[1][v] - mnew);
      S[0][v] = p0; S[1][v] = p1;
      float rs = p0 + p1;
#pragma unroll
      for (int off = 8; off > 0; off >>= 1)
        rs += __shfl_xor(rs, off, 32);
      lrun[v] = lrun[v] * corr + rs;
      mrun[v] = mnew;
#pragma unroll
      for (int nt = 0; nt < 4; ++nt) o_acc[nt][v] *= corr;
    }
    // P: C layout -> row-major 16x32 in per-wave LDS, re-read as A fragment
#pragma unroll
    for (int v = 0; v < 8; ++v) {
      pl[(v + 8 * half) * 32 + lm]      = f2bf(S[0][v]);
      pl[(v + 8 * half) * 32 + 16 + lm] = f2bf(S[1][v]);
    }
    asm volatile("s_wait_dscnt 0" ::: "memory");    // wave-private RAW on LDS
    v16bf pa = load_A_frag_k1(pl, 32);
#pragma unroll
    for (int nt = 0; nt < 4; ++nt) {
      v16bf vb = load_B_frag_k1(vtp + (size_t)(nt * 16) * Tz + key0, Tz);
      o_acc[nt] = __builtin_amdgcn_wmma_f32_16x16x32_bf16(
          false, pa, false, vb, (short)0, o_acc[nt], false, false);
    }
  };

  // full (provably unmasked) tiles, then one masked boundary tile
  int kfull = q0 & ~31;
  for (int key0 = 0; key0 < kfull; key0 += 32) process_tile(key0, false);
  process_tile(kfull, true);

  // normalize + store bf16 output (head-interleaved [B*T, E])
  float inv[8];
#pragma unroll
  for (int v = 0; v < 8; ++v) inv[v] = __builtin_amdgcn_rcpf(lrun[v]);
#pragma unroll
  for (int nt = 0; nt < 4; ++nt)
#pragma unroll
    for (int v = 0; v < 8; ++v) {
      float val = o_acc[nt][v] * inv[v];
      O[base + (size_t)(q0 + v + 8 * half) * Ez + nt * 16 + lm] = f2bf(val);
    }
}

// ---------- host orchestration ----------
extern "C" void kernel_launch(void* const* d_in, const int* in_sizes, int n_in,
                              void* d_out, int out_size, void* d_ws, size_t ws_size,
                              hipStream_t stream) {
  (void)in_sizes; (void)n_in; (void)out_size; (void)ws_size;
  const float* x    = (const float*)d_in[0];
  const float* ln1g = (const float*)d_in[1];
  const float* ln1b = (const float*)d_in[2];
  const float* Wq   = (const float*)d_in[3];
  const float* Wk   = (const float*)d_in[4];
  const float* Wv   = (const float*)d_in[5];
  const float* Wo   = (const float*)d_in[6];
  const float* bo   = (const float*)d_in[7];
  const float* ln2g = (const float*)d_in[8];
  const float* ln2b = (const float*)d_in[9];
  const float* W1   = (const float*)d_in[10];
  const float* b1   = (const float*)d_in[11];
  const float* W2   = (const float*)d_in[12];
  const float* b2   = (const float*)d_in[13];
  float* out = (float*)d_out;

  char* ws = (char*)d_ws;
  size_t off = 0;
  auto alloc = [&](size_t bytes) -> char* {
    char* p = ws + off;
    off += (bytes + 255) & ~(size_t)255;
    return p;
  };
  // activations (bf16 unless noted); K padded 32 rows, vt padded 64 elems
  __bf16* ln1bf  = (__bf16*)alloc((size_t)Mz * Ez * 2);
  __bf16* qbf    = (__bf16*)alloc(((size_t)Mz + 32) * Ez * 2);
  __bf16* kbf    = (__bf16*)alloc(((size_t)Mz + 32) * Ez * 2);
  __bf16* vbf    = (__bf16*)alloc(((size_t)Mz + 32) * Ez * 2);
  __bf16* vt     = (__bf16*)alloc(((size_t)Bz * Hz * Dz * Tz + 64) * 2);
  __bf16* attnbf = (__bf16*)alloc((size_t)Mz * Ez * 2);
  float*  x1     = (float*) alloc((size_t)Mz * Ez * 4);   // residual stream
  __bf16* hbf    = (__bf16*)alloc((size_t)Mz * Ez * 2);
  __bf16* h1bf   = (__bf16*)alloc((size_t)Mz * FF * 2);
  // transposed bf16 weights Wt[n][k]
  __bf16* Wqt = (__bf16*)alloc((size_t)Ez * Ez * 2);
  __bf16* Wkt = (__bf16*)alloc((size_t)Ez * Ez * 2);
  __bf16* Wvt = (__bf16*)alloc((size_t)Ez * Ez * 2);
  __bf16* Wot = (__bf16*)alloc((size_t)Ez * Ez * 2);
  __bf16* W1t = (__bf16*)alloc((size_t)Ez * FF * 2);
  __bf16* W2t = (__bf16*)alloc((size_t)FF * Ez * 2);

  // weights: fp32 [K][N] -> bf16 transposed [N][K]
  cvt_transpose_kernel<<<(Ez * Ez) / 256, 256, 0, stream>>>(Wq, Wqt, 10, Ez);
  cvt_transpose_kernel<<<(Ez * Ez) / 256, 256, 0, stream>>>(Wk, Wkt, 10, Ez);
  cvt_transpose_kernel<<<(Ez * Ez) / 256, 256, 0, stream>>>(Wv, Wvt, 10, Ez);
  cvt_transpose_kernel<<<(Ez * Ez) / 256, 256, 0, stream>>>(Wo, Wot, 10, Ez);
  cvt_transpose_kernel<<<(Ez * FF) / 256, 256, 0, stream>>>(W1, W1t, 10, FF);
  cvt_transpose_kernel<<<(FF * Ez) / 256, 256, 0, stream>>>(W2, W2t, 12, Ez);

  dim3 gE(Ez / 128, Mz / 128);     // N=1024 outputs
  dim3 gF(FF / 128, Mz / 128);     // N=4096 outputs

  // 1) LN1 -> bf16
  ln_bf16_kernel<<<Mz, 256, 0, stream>>>(x, ln1g, ln1b, ln1bf);
  // 2) Q/K/V projections (bf16 out); Q is pre-scaled by 1/sqrt(D)
  gemm_bf16_kernel<Ez, false, true, false, false, false, true>
      <<<gE, 256, 0, stream>>>(ln1bf, Wqt, nullptr, nullptr, nullptr, qbf, Ez);
  gemm_bf16_kernel<Ez, false, true, false, false, false, false>
      <<<gE, 256, 0, stream>>>(ln1bf, Wkt, nullptr, nullptr, nullptr, kbf, Ez);
  gemm_bf16_kernel<Ez, false, true, false, false, false, false>
      <<<gE, 256, 0, stream>>>(ln1bf, Wvt, nullptr, nullptr, nullptr, vbf, Ez);
  // 2b) transpose V for key-contiguous PV fragments
  transpose_v_kernel<<<(Bz * Hz * Dz * Tz) / 256, 256, 0, stream>>>(vbf, vt);
  // 3) causal flash attention (4096 waves, 8 per block)
  attn_fwd_kernel<<<512, 256, 0, stream>>>(qbf, kbf, vt, attnbf);
  // 4) output projection + bo + residual(x) -> x1 (fp32)
  gemm_bf16_kernel<Ez, true, false, false, true, true, false>
      <<<gE, 256, 0, stream>>>(attnbf, Wot, bo, x, x1, nullptr, Ez);
  // 5) LN2 -> bf16
  ln_bf16_kernel<<<Mz, 256, 0, stream>>>(x1, ln2g, ln2b, hbf);
  // 6) MLP up + b1 + relu (bf16 out)
  gemm_bf16_kernel<Ez, false, true, true, false, true, false>
      <<<gF, 256, 0, stream>>>(hbf, W1t, b1, nullptr, nullptr, h1bf, FF);
  // 7) MLP down + b2 + residual(x1) -> out (fp32)
  gemm_bf16_kernel<FF, true, false, false, true, true, false>
      <<<gE, 256, 0, stream>>>(h1bf, W2t, b2, x1, out, nullptr, Ez);
}
